// IdentificationLossInBatch_26843545600183
// MI455X (gfx1250) — matrix operand order repaired and checked by reading, not compiled, not run on hardware
//
#include <hip/hip_runtime.h>
#include <math.h>

typedef __attribute__((ext_vector_type(16))) __bf16 v16bf;
typedef __attribute__((ext_vector_type(8)))  float  v8f;
typedef int v4i_vs __attribute__((vector_size(16)));
typedef __attribute__((address_space(1))) v4i_vs* gv4i_p;   // global int4*
typedef __attribute__((address_space(3))) v4i_vs* lv4i_p;   // LDS int4*
typedef unsigned int u32x4 __attribute__((ext_vector_type(4)));
typedef int          i32x8 __attribute__((ext_vector_type(8)));
typedef int          i32x4 __attribute__((ext_vector_type(4)));

#define NROW 8192
#define KDIM 1024
#define BM 32
#define BN 256
#define KC 32
#define NSTEPS ((NROW / BN) * (KDIM / KC))    // 1024 flat B tiles
#define SA_STRIDE 1032       // ushorts: 1024 + 8 pad -> 2064B rows, 16B aligned, bank-staggered
#define SB_STRIDE 40         // ushorts: 32 + 8 pad   -> 80B rows (TDM pad reproduces this)
#define B_HALF_USH (BN * SB_STRIDE)           // 10240 ushorts per parity buffer
#define B_HALF_BYTES (B_HALF_USH * 2)         // 20480
#define INV_T (1.0f / 0.07f)

// LDS carve (bytes)
#define OFF_AHI   0
#define OFF_ALO   66048                       // 32*1032*2
#define OFF_BHI   132096                      // 2 x 20480 (double buffered)
#define OFF_BLO   173056
#define OFF_STATS 214016                      // 8*32*2 floats
#define OFF_POS   216064                      // 32 floats
#define SMEM_BYTES 216192

// workspace layout (bytes)
#define WS_PART 0
#define WS_QHI  4096
#define WS_QLO  (4096 + NROW * KDIM * 2)

#if __has_builtin(__builtin_amdgcn_tensor_load_to_lds) && \
    __has_builtin(__builtin_amdgcn_s_wait_tensorcnt)
#define USE_TDM 1
#elif __has_builtin(__builtin_amdgcn_global_load_async_to_lds_b128) && \
      __has_builtin(__builtin_amdgcn_s_wait_asynccnt)
#define USE_ASYNC_LDS 1
#endif

struct bfpack { uint4 a; uint4 b; };
__device__ __forceinline__ v16bf ld_op(const unsigned short* base, int off0, int off1) {
  bfpack p;
  p.a = *(const uint4*)(base + off0);
  p.b = *(const uint4*)(base + off1);
  return __builtin_bit_cast(v16bf, p);
}

__device__ __forceinline__ unsigned int lds_off_of(const void* p) {
  return (unsigned int)(uintptr_t)(__attribute__((address_space(3))) const void*)p;
}

#ifdef USE_TDM
// One 2D TDM tile: 256 rows x 64B from a row-major bf16 matrix (row stride 2048B),
// LDS-padded +16B per 64B row -> 80B LDS row stride (== SB_STRIDE ushorts).
__device__ __forceinline__ void tdm_issue(const unsigned short* gsrc, unsigned int lds_byte_off) {
  unsigned long long ga = (unsigned long long)(uintptr_t)gsrc;
  u32x4 g0 = {
    1u,                                                   // count=1 (valid user descriptor)
    lds_byte_off,                                         // lds_addr
    (unsigned int)ga,                                     // global_addr[31:0]
    ((unsigned int)(ga >> 32) & 0x1FFFFFFu) | (2u << 30)  // global_addr[56:32] | type=2
  };
  i32x8 g1 = {
    (int)0x06D30000,   // data_size=3(8B) | pad_enable | pad_interval=3(16DW) | pad_amount=3(4DW)
    (int)0x01000000,   // tensor_dim0 = 256 (8B units) -> bits[63:48]
    (int)0x20000000,   // tensor_dim1 = 8192 rows      -> bits[95:80]
    (int)0x00080000,   // tile_dim0 = 8 (8B units = 64B) -> bits[127:112]
    256,               // tile_dim1 = 256 rows
    256,               // tensor_dim0_stride = 256 (8B units = 2048B)
    0, 0
  };
  i32x4 gz4 = {0, 0, 0, 0};
  i32x8 gz8 = {0, 0, 0, 0, 0, 0, 0, 0};
  __builtin_amdgcn_tensor_load_to_lds(g0, g1, gz4, gz4, gz8, 0);
}
#endif

#ifndef USE_TDM
// cooperative copy of tile (jn,kn) into parity buffer npar (async or plain)
__device__ __forceinline__ void stage_copy(const unsigned short* __restrict__ qhi,
                                           const unsigned short* __restrict__ qlo,
                                           unsigned short* bHi, unsigned short* bLo,
                                           int jn, int kn, int npar, int t) {
#pragma unroll
  for (int u = 0; u < 4; ++u) {
    int idx  = t * 4 + u;                   // 0..1023
    int brow = idx >> 2;                    // 0..255 (4 uint4 per row)
    int c8   = (idx & 3) * 8;
    const unsigned short* sh = qhi + (size_t)(jn + brow) * KDIM + kn + c8;
    const unsigned short* sl = qlo + (size_t)(jn + brow) * KDIM + kn + c8;
    unsigned short* dh = bHi + npar * B_HALF_USH + brow * SB_STRIDE + c8;
    unsigned short* dl = bLo + npar * B_HALF_USH + brow * SB_STRIDE + c8;
#ifdef USE_ASYNC_LDS
    __builtin_amdgcn_global_load_async_to_lds_b128((gv4i_p)sh, (lv4i_p)dh, 0, 0);
    __builtin_amdgcn_global_load_async_to_lds_b128((gv4i_p)sl, (lv4i_p)dl, 0, 0);
#else
    *(uint4*)dh = *(const uint4*)sh;
    *(uint4*)dl = *(const uint4*)sl;
#endif
  }
}
#endif

// ---------------- pre-pass: q (fp32) -> q_hi + q_lo (bf16 split), done ONCE ----------------
__global__ void __launch_bounds__(256, 1)
ident_split_pre(const float* __restrict__ q,
                unsigned short* __restrict__ qhi, unsigned short* __restrict__ qlo) {
  size_t base = ((size_t)blockIdx.x * 256 + threadIdx.x) * 8;
  const float4* q4 = (const float4*)(q + base);
  float4 a = q4[0], b = q4[1];
  float xs[8] = {a.x, a.y, a.z, a.w, b.x, b.y, b.z, b.w};
  unsigned int hp[4], lp[4];
#pragma unroll
  for (int p = 0; p < 4; ++p) {
    unsigned int u0 = __float_as_uint(xs[2 * p]);
    unsigned int u1 = __float_as_uint(xs[2 * p + 1]);
    hp[p] = (u0 >> 16) | (u1 & 0xFFFF0000u);
    float r0 = xs[2 * p]     - __uint_as_float(u0 & 0xFFFF0000u);
    float r1 = xs[2 * p + 1] - __uint_as_float(u1 & 0xFFFF0000u);
    lp[p] = (__float_as_uint(r0) >> 16) | (__float_as_uint(r1) & 0xFFFF0000u);
  }
  uint4 h; h.x = hp[0]; h.y = hp[1]; h.z = hp[2]; h.w = hp[3];
  uint4 l; l.x = lp[0]; l.y = lp[1]; l.z = lp[2]; l.w = lp[3];
  *(uint4*)(qhi + base) = h;
  *(uint4*)(qlo + base) = l;
}

// ---------------- main fused GEMM + online logsumexp ----------------
__global__ void __launch_bounds__(256, 1)
ident_loss_main(const float* __restrict__ q,
                const unsigned short* __restrict__ qhi,
                const unsigned short* __restrict__ qlo,
                float* __restrict__ part) {
  extern __shared__ __align__(16) char smem[];
  unsigned short* aHi = (unsigned short*)(smem + OFF_AHI);
  unsigned short* aLo = (unsigned short*)(smem + OFF_ALO);
  unsigned short* bHi = (unsigned short*)(smem + OFF_BHI);
  unsigned short* bLo = (unsigned short*)(smem + OFF_BLO);
  float* sStats = (float*)(smem + OFF_STATS);   // [8][32][2]
  float* sPos   = (float*)(smem + OFF_POS);     // [32]

  const int t    = threadIdx.x;
  const int lane = t & 31;
  const int wave = t >> 5;           // 0..7 : column group (cols wave*16 and 128+wave*16)
  const int half = lane >> 4;
  const int l4   = lane & 15;
  const int row0 = blockIdx.x * BM;

  // ---- stage A (bf16 hi/lo, full K) into LDS: pure b128 copies ----
  for (int i = 0; i < 16; ++i) {
    int idx = i * 256 + t;
    int row = idx >> 7;
    int c8  = (idx & 127) * 8;
    *(uint4*)(aHi + row * SA_STRIDE + c8) = *(const uint4*)(qhi + (size_t)(row0 + row) * KDIM + c8);
    *(uint4*)(aLo + row * SA_STRIDE + c8) = *(const uint4*)(qlo + (size_t)(row0 + row) * KDIM + c8);
  }

  // ---- pos_i = <q_i,q_i>/T, exact fp32 ----
  {
    int prow = t >> 3, c = t & 7;
    const float4* q4 = (const float4*)q;
    float acc = 0.f;
    for (int i = 0; i < 32; ++i) {
      float4 v = q4[(size_t)(row0 + prow) * (KDIM / 4) + c * 32 + i];
      acc += v.x * v.x + v.y * v.y + v.z * v.z + v.w * v.w;
    }
    acc += __shfl_xor(acc, 1, 32);
    acc += __shfl_xor(acc, 2, 32);
    acc += __shfl_xor(acc, 4, 32);
    if (c == 0) sPos[prow] = acc * INV_T;
  }

  // per-lane private online softmax state: 4 tiles = {mi0,mi1} x {ni0,ni1}
  float mm[4][8], ss[4][8];
#pragma unroll
  for (int c = 0; c < 4; ++c)
#pragma unroll
    for (int r = 0; r < 8; ++r) { mm[c][r] = -INFINITY; ss[c][r] = 0.f; }

  const int am0 = l4;                 // A rows 0-15
  const int am1 = 16 + l4;            // A rows 16-31
  const int bn0 = 16 * wave + l4;     // B row (= S column), first n-tile
  const int bn1 = bn0 + 128;          // second n-tile

#ifdef USE_TDM
  const unsigned int ldsBHi = lds_off_of(bHi);
  const unsigned int ldsBLo = lds_off_of(bLo);
#endif

  // ---- pipeline prologue: stage flat tile 0 into parity 0 ----
#ifdef USE_TDM
  if (wave == 0) {
    tdm_issue(qhi, ldsBHi);
    tdm_issue(qlo, ldsBLo);
  }
#else
  stage_copy(qhi, qlo, bHi, bLo, 0, 0, 0, t);
#endif

  int s = 0, par = 0;
  int jn = 0, kn = KC;          // coordinates of flat tile s+1

  for (int j0 = 0; j0 < NROW; j0 += BN) {
    v8f acc[4];
#pragma unroll
    for (int c = 0; c < 4; ++c) acc[c] = (v8f){0.f, 0.f, 0.f, 0.f, 0.f, 0.f, 0.f, 0.f};

    for (int k0 = 0; k0 < KDIM; k0 += KC) {
      __syncthreads();                       // all reads of buf[par^1] (step s-1) done
      const bool have_next = (s + 1 < NSTEPS);
      if (have_next) {
#ifdef USE_TDM
        if (wave == 0) {
          const unsigned short* gh = qhi + (size_t)jn * KDIM + kn;
          const unsigned short* gl = qlo + (size_t)jn * KDIM + kn;
          tdm_issue(gh, ldsBHi + (unsigned)((par ^ 1) * B_HALF_BYTES));
          tdm_issue(gl, ldsBLo + (unsigned)((par ^ 1) * B_HALF_BYTES));
        }
#else
        stage_copy(qhi, qlo, bHi, bLo, jn, kn, par ^ 1, t);
#endif
      }
#ifdef USE_TDM
      if (have_next) __builtin_amdgcn_s_wait_tensorcnt(2);   // tile s done; s+1 in flight
      else           __builtin_amdgcn_s_wait_tensorcnt(0);
#elif defined(USE_ASYNC_LDS)
      if (have_next) __builtin_amdgcn_s_wait_asynccnt(8);    // tile s done; s+1 in flight
      else           __builtin_amdgcn_s_wait_asynccnt(0);
#endif
      __syncthreads();                       // buf[par] visible to all waves

      // operands (ISA VGPR layouts)
      const unsigned short* pbH = bHi + par * B_HALF_USH;
      const unsigned short* pbL = bLo + par * B_HALF_USH;
      v16bf a_h0 = ld_op(aHi + am0 * SA_STRIDE + k0, 8 * half, 16 + 8 * half);
      v16bf a_l0 = ld_op(aLo + am0 * SA_STRIDE + k0, 8 * half, 16 + 8 * half);
      v16bf a_h1 = ld_op(aHi + am1 * SA_STRIDE + k0, 8 * half, 16 + 8 * half);
      v16bf a_l1 = ld_op(aLo + am1 * SA_STRIDE + k0, 8 * half, 16 + 8 * half);
      v16bf b_h0 = ld_op(pbH + bn0 * SB_STRIDE, 8 * half, 16 + 8 * half);
      v16bf b_l0 = ld_op(pbL + bn0 * SB_STRIDE, 8 * half, 16 + 8 * half);
      v16bf b_h1 = ld_op(pbH + bn1 * SB_STRIDE, 8 * half, 16 + 8 * half);
      v16bf b_l1 = ld_op(pbL + bn1 * SB_STRIDE, 8 * half, 16 + 8 * half);

      // fp32-split (hi*hi + hi*lo + lo*hi) on a 2x2 tile grid: 12 WMMAs, 16 ds loads
      acc[0] = __builtin_amdgcn_wmma_f32_16x16x32_bf16(false, a_h0, false, b_h0, (short)0, acc[0], false, false);
      acc[1] = __builtin_amdgcn_wmma_f32_16x16x32_bf16(false, a_h0, false, b_h1, (short)0, acc[1], false, false);
      acc[2] = __builtin_amdgcn_wmma_f32_16x16x32_bf16(false, a_h1, false, b_h0, (short)0, acc[2], false, false);
      acc[3] = __builtin_amdgcn_wmma_f32_16x16x32_bf16(false, a_h1, false, b_h1, (short)0, acc[3], false, false);
      acc[0] = __builtin_amdgcn_wmma_f32_16x16x32_bf16(false, a_h0, false, b_l0, (short)0, acc[0], false, false);
      acc[1] = __builtin_amdgcn_wmma_f32_16x16x32_bf16(false, a_h0, false, b_l1, (short)0, acc[1], false, false);
      acc[2] = __builtin_amdgcn_wmma_f32_16x16x32_bf16(false, a_h1, false, b_l0, (short)0, acc[2], false, false);
      acc[3] = __builtin_amdgcn_wmma_f32_16x16x32_bf16(false, a_h1, false, b_l1, (short)0, acc[3], false, false);
      acc[0] = __builtin_amdgcn_wmma_f32_16x16x32_bf16(false, a_l0, false, b_h0, (short)0, acc[0], false, false);
      acc[1] = __builtin_amdgcn_wmma_f32_16x16x32_bf16(false, a_l0, false, b_h1, (short)0, acc[1], false, false);
      acc[2] = __builtin_amdgcn_wmma_f32_16x16x32_bf16(false, a_l1, false, b_h0, (short)0, acc[2], false, false);
      acc[3] = __builtin_amdgcn_wmma_f32_16x16x32_bf16(false, a_l1, false, b_h1, (short)0, acc[3], false, false);

      par ^= 1; ++s;
      kn += KC; if (kn == KDIM) { kn = 0; jn += BN; }
    }

    // ---- per-lane online logsumexp update (registers only) ----
#pragma unroll
    for (int c = 0; c < 4; ++c) {
#pragma unroll
      for (int r = 0; r < 8; ++r) {
        float x = acc[c][r] * INV_T;
        float n = fmaxf(mm[c][r], x);
        ss[c][r] = ss[c][r] * __expf(mm[c][r] - n) + __expf(x - n);
        mm[c][r] = n;
      }
    }
  }

  // ---- end-only merges: ni pair -> 16 lanes -> 8 waves ----
  __syncthreads();
#pragma unroll
  for (int mi = 0; mi < 2; ++mi) {
#pragma unroll
    for (int r = 0; r < 8; ++r) {
      int c0 = 2 * mi, c1 = 2 * mi + 1;           // same rows, two column groups
      float M = fmaxf(mm[c0][r], mm[c1][r]);
      float S = ss[c0][r] * __expf(mm[c0][r] - M) + ss[c1][r] * __expf(mm[c1][r] - M);
#pragma unroll
      for (int mask = 1; mask <= 8; mask <<= 1) {
        float Mo = __shfl_xor(M, mask, 32);
        float So = __shfl_xor(S, mask, 32);
        float Mn = fmaxf(M, Mo);
        S = S * __expf(M - Mn) + So * __expf(Mo - Mn);
        M = Mn;
      }
      if (l4 == 0) {
        int mrow = 16 * mi + r + 8 * half;        // 0..31
        sStats[(wave * 32 + mrow) * 2 + 0] = M;
        sStats[(wave * 32 + mrow) * 2 + 1] = S;
      }
    }
  }
  __syncthreads();
  if (t < 32) {
    float M = -INFINITY;
#pragma unroll
    for (int w = 0; w < 8; ++w) M = fmaxf(M, sStats[(w * 32 + t) * 2 + 0]);
    float S = 0.f;
#pragma unroll
    for (int w = 0; w < 8; ++w) S += sStats[(w * 32 + t) * 2 + 1] * __expf(sStats[(w * 32 + t) * 2 + 0] - M);
    float loss = logf(S) + M - sPos[t];
    loss += __shfl_xor(loss, 1, 32);
    loss += __shfl_xor(loss, 2, 32);
    loss += __shfl_xor(loss, 4, 32);
    loss += __shfl_xor(loss, 8, 32);
    loss += __shfl_xor(loss, 16, 32);
    if (t == 0) part[blockIdx.x] = loss;
  }
}

__global__ void __launch_bounds__(256, 1)
ident_loss_finalize(const float* __restrict__ part, float* __restrict__ out) {
  __shared__ float sw[8];
  int t = threadIdx.x;
  float v = part[t];
  v += __shfl_xor(v, 1, 32);
  v += __shfl_xor(v, 2, 32);
  v += __shfl_xor(v, 4, 32);
  v += __shfl_xor(v, 8, 32);
  v += __shfl_xor(v, 16, 32);
  if ((t & 31) == 0) sw[t >> 5] = v;
  __syncthreads();
  if (t == 0) {
    float s = 0.f;
    for (int i = 0; i < 8; ++i) s += sw[i];
    out[0] = s * (1.0f / (float)NROW);
  }
}

extern "C" void kernel_launch(void* const* d_in, const int* in_sizes, int n_in,
                              void* d_out, int out_size, void* d_ws, size_t ws_size,
                              hipStream_t stream) {
  (void)in_sizes; (void)n_in; (void)out_size; (void)ws_size;
  const float* q = (const float*)d_in[0];
  float* out  = (float*)d_out;
  float* part = (float*)((char*)d_ws + WS_PART);
  unsigned short* qhi = (unsigned short*)((char*)d_ws + WS_QHI);
  unsigned short* qlo = (unsigned short*)((char*)d_ws + WS_QLO);

  ident_split_pre<<<(NROW * KDIM) / (256 * 8), 256, 0, stream>>>(q, qhi, qlo);

  (void)hipFuncSetAttribute((const void*)ident_loss_main,
                            hipFuncAttributeMaxDynamicSharedMemorySize, SMEM_BYTES);
  ident_loss_main<<<NROW / BM, 256, SMEM_BYTES, stream>>>(q, qhi, qlo, part);
  ident_loss_finalize<<<1, 256, 0, stream>>>(part, out);
}
